// NodeContrastiveLoss_10917806866571
// MI455X (gfx1250) — matrix operand order repaired and checked by reading, not compiled, not run on hardware
//
#include <hip/hip_runtime.h>
#include <hip/hip_fp16.h>

// ---------------------------------------------------------------------------
// Node contrastive loss (InfoNCE, tau=0.07) for N=16384, D=128 fp32 inputs.
//
// Compute-bound (137 GFLOP vs 16 MB input, all L2-resident). Pipeline:
//  1) normalize rows -> f16 copies, pos[i] = dot(z1n,z2n)/tau, and
//     selfexp[i] = exp(||z1h_i||^2 / tau) (the f16-rounded self-similarity,
//     used to remove the z1@z1 diagonal in the epilogue instead of masking
//     inside the hot loop).
//  2) streaming sim kernel: each wave owns TWO 16-row i-tiles (shared B
//     loads, 8 v_wmma_f32_16x16x32_f16 per 8 global_load_b128) and strides
//     the 2N/16 j-tiles, folding exp2(sim * log2e/tau) into per-lane
//     accumulators via the RAW v_exp_f32 (args bounded in [-20.7, 20.7],
//     so no OCML subnormal-range fixup needed).
//  3) epilogue: loss = mean(log(sumexp - selfexp_correction) - pos).
// exp never overflows fp32 since sim/tau <= 14.3 (sum <= ~5e10).
// ---------------------------------------------------------------------------

#define D_DIM 128
#define TAU_INV 14.285714285714286f              // 1 / 0.07
#define EXP2_SCALE 20.609929155512072f           // log2(e) / 0.07

typedef _Float16 half4_t  __attribute__((ext_vector_type(4)));
typedef _Float16 half8_t  __attribute__((ext_vector_type(8)));
typedef _Float16 half16_t __attribute__((ext_vector_type(16)));
typedef float    float8_t __attribute__((ext_vector_type(8)));

// Raw v_exp_f32: valid because |x| <= ~21 in the hot loop (no subnormal /
// overflow range handling required, unlike OCML exp2f).
__device__ __forceinline__ float fast_exp2(float x) {
#if __has_builtin(__builtin_amdgcn_exp2f)
    return __builtin_amdgcn_exp2f(x);
#else
    return exp2f(x);
#endif
}

// -------------------------------------------------------------------------
// Kernel 1: per-row L2 normalize both views, emit f16 copies, pos[i], and
// selfexp[i]. One wave32 per row; each lane owns 4 contiguous floats.
// -------------------------------------------------------------------------
__global__ __launch_bounds__(256) void ncl_normalize_kernel(
    const float* __restrict__ z1, const float* __restrict__ z2,
    _Float16* __restrict__ z1h, _Float16* __restrict__ z2h,
    float* __restrict__ pos, float* __restrict__ selfexp, int N)
{
    const int lane = threadIdx.x & 31;
    const int row  = (int)((blockIdx.x * blockDim.x + threadIdx.x) >> 5);
    if (row >= N) return;

    const float4 a = ((const float4*)(z1 + (size_t)row * D_DIM))[lane];
    const float4 b = ((const float4*)(z2 + (size_t)row * D_DIM))[lane];

    float sa = a.x * a.x + a.y * a.y + a.z * a.z + a.w * a.w;
    float sb = b.x * b.x + b.y * b.y + b.z * b.z + b.w * b.w;
#pragma unroll
    for (int m = 16; m >= 1; m >>= 1) {
        sa += __shfl_xor(sa, m, 32);
        sb += __shfl_xor(sb, m, 32);
    }
    const float ra = 1.0f / fmaxf(sqrtf(sa), 1e-12f);
    const float rb = 1.0f / fmaxf(sqrtf(sb), 1e-12f);

    const float ax = a.x * ra, ay = a.y * ra, az = a.z * ra, aw = a.w * ra;
    const float bx = b.x * rb, by = b.y * rb, bz = b.z * rb, bw = b.w * rb;

    // f16-rounded values (what the WMMA pipeline will actually see).
    const _Float16 hax = (_Float16)ax, hay = (_Float16)ay,
                   haz = (_Float16)az, haw = (_Float16)aw;
    const _Float16 hbx = (_Float16)bx, hby = (_Float16)by,
                   hbz = (_Float16)bz, hbw = (_Float16)bw;

    // pos uses the full-precision normalized vectors (matches reference).
    float dot = ax * bx + ay * by + az * bz + aw * bw;
    // self-similarity of the f16-rounded z1 row (f16^2 is exact in f32).
    const float fax = (float)hax, fay = (float)hay,
                faz = (float)haz, faw = (float)haw;
    float qa = fax * fax + fay * fay + faz * faz + faw * faw;
#pragma unroll
    for (int m = 16; m >= 1; m >>= 1) {
        dot += __shfl_xor(dot, m, 32);
        qa  += __shfl_xor(qa, m, 32);
    }
    if (lane == 0) {
        pos[row]     = dot * TAU_INV;
        selfexp[row] = exp2f(qa * EXP2_SCALE);
    }

    half4_t ha = { hax, hay, haz, haw };
    half4_t hb = { hbx, hby, hbz, hbw };
    *(half4_t*)(z1h + (size_t)row * D_DIM + lane * 4) = ha;
    *(half4_t*)(z2h + (size_t)row * D_DIM + lane * 4) = hb;
}

// -------------------------------------------------------------------------
// Kernel 2: streaming sim + exp accumulation.
// Block = 8 waves; block b owns rows [32b, 32b+32) as two 16-row i-tiles.
// Each wave strides the N/16 j-tiles of z2, then of z1 (two uniform loops,
// no per-iteration operand select). Per j-tile: 8 global_load_b128 B
// fragments feed 8 WMMAs (2 i-tiles x 4 K-chunks), then 16 exp2+add.
//
// Fragment layouts per CDNA5 ISA 7.12.2 (wave32):
//   A 16x32 f16 : lane l (m=l&15, h=l>>4) holds row ibase+m,
//                 halves k0+8h..+7 then k0+16+8h..+7.
//   B 32x16 f16 : lane l (n=l&15, h=l>>4) holds column jbase+n,
//                 halves k0+8h..+7 then k0+16+8h..+7.
//   C/D f32     : vgpr r, lane l -> element (row r+8h, col n).
// -------------------------------------------------------------------------
__global__ __launch_bounds__(256) void ncl_sim_kernel(
    const _Float16* __restrict__ z1h, const _Float16* __restrict__ z2h,
    float* __restrict__ sumexp, int N)
{
    const int lane   = threadIdx.x & 31;
    const int wave   = threadIdx.x >> 5;
    const int nwaves = blockDim.x >> 5;   // 8
    const int ibase  = blockIdx.x * 32;   // two 16-row i-tiles
    const int q      = lane & 15;         // row (A) / col (B) within tile
    const int h      = lane >> 4;         // half-wave select

    // Preload A fragments for both i-tiles, all four K-chunks (reused 2048x).
    half16_t afrag[2][4];
#pragma unroll
    for (int t = 0; t < 2; ++t) {
        const _Float16* arow = z1h + (size_t)(ibase + t * 16 + q) * D_DIM;
#pragma unroll
        for (int kk = 0; kk < 4; ++kk) {
            half8_t lo = *(const half8_t*)(arow + kk * 32 + h * 8);
            half8_t hi = *(const half8_t*)(arow + kk * 32 + 16 + h * 8);
            afrag[t][kk] = __builtin_shufflevector(lo, hi, 0, 1, 2, 3, 4, 5, 6,
                                                   7, 8, 9, 10, 11, 12, 13, 14,
                                                   15);
        }
    }

    float accs[2][8];
#pragma unroll
    for (int t = 0; t < 2; ++t)
#pragma unroll
        for (int r = 0; r < 8; ++r) accs[t][r] = 0.0f;

    const int ntiles = N >> 4;
    const unsigned step = (unsigned)nwaves * 16u * D_DIM;  // elems per stride

#pragma unroll 1
    for (int pass = 0; pass < 2; ++pass) {
        const _Float16* zb = pass ? z1h : z2h;
        unsigned off = (unsigned)(wave * 16 + q) * D_DIM + (unsigned)h * 8u;
#pragma unroll 1
        for (int jt = wave; jt < ntiles; jt += nwaves, off += step) {
            float8_t c0 = {}, c1 = {};
#pragma unroll
            for (int kk = 0; kk < 4; ++kk) {
                half8_t lo = *(const half8_t*)(zb + off + kk * 32);
                half8_t hi = *(const half8_t*)(zb + off + kk * 32 + 16);
                half16_t bfrag = __builtin_shufflevector(
                    lo, hi, 0, 1, 2, 3, 4, 5, 6, 7, 8, 9, 10, 11, 12, 13, 14,
                    15);
                c0 = __builtin_amdgcn_wmma_f32_16x16x32_f16(
                    false, afrag[0][kk], false, bfrag, (short)0, c0, false,
                    false);
                c1 = __builtin_amdgcn_wmma_f32_16x16x32_f16(
                    false, afrag[1][kk], false, bfrag, (short)0, c1, false,
                    false);
            }
#pragma unroll
            for (int r = 0; r < 8; ++r) {
                accs[0][r] += fast_exp2(c0[r] * EXP2_SCALE);
                accs[1][r] += fast_exp2(c1[r] * EXP2_SCALE);
            }
        }
    }

    // Reduce each row across the 16 lanes of its half-wave, then one lane
    // per (row, half) commits the partial to global sumexp.
#pragma unroll
    for (int t = 0; t < 2; ++t) {
#pragma unroll
        for (int r = 0; r < 8; ++r) {
            float v = accs[t][r];
            v += __shfl_xor(v, 1, 32);
            v += __shfl_xor(v, 2, 32);
            v += __shfl_xor(v, 4, 32);
            v += __shfl_xor(v, 8, 32);
            if (q == r) atomicAdd(&sumexp[ibase + t * 16 + r + 8 * h], v);
        }
    }
}

// -------------------------------------------------------------------------
// Kernel 3: loss = mean(log(sumexp - selfexp) - pos).
// -------------------------------------------------------------------------
__global__ __launch_bounds__(256) void ncl_loss_kernel(
    const float* __restrict__ pos, const float* __restrict__ sumexp,
    const float* __restrict__ selfexp, float* __restrict__ out, int N)
{
    __shared__ float sm[256];
    float local = 0.0f;
    for (int i = (int)(blockIdx.x * blockDim.x + threadIdx.x); i < N;
         i += (int)(gridDim.x * blockDim.x))
        local += logf(sumexp[i] - selfexp[i]) - pos[i];
    sm[threadIdx.x] = local;
    __syncthreads();
#pragma unroll
    for (int s = 128; s > 0; s >>= 1) {
        if ((int)threadIdx.x < s) sm[threadIdx.x] += sm[threadIdx.x + s];
        __syncthreads();
    }
    if (threadIdx.x == 0) atomicAdd(out, sm[0] / (float)N);
}

extern "C" void kernel_launch(void* const* d_in, const int* in_sizes, int n_in,
                              void* d_out, int out_size, void* d_ws, size_t ws_size,
                              hipStream_t stream) {
    const float* z1 = (const float*)d_in[0];
    const float* z2 = (const float*)d_in[1];
    const int N = in_sizes[0] / D_DIM;

    // ws: z1h (N*D*2B) | z2h (N*D*2B) | pos (N*4B) | sumexp (N*4B) | selfexp (N*4B)
    char* ws = (char*)d_ws;
    _Float16* z1h  = (_Float16*)ws;
    _Float16* z2h  = (_Float16*)(ws + (size_t)N * D_DIM * 2);
    float* pos     = (float*)(ws + (size_t)N * D_DIM * 4);
    float* sumexp  = (float*)(ws + (size_t)N * D_DIM * 4 + (size_t)N * 4);
    float* selfexp = (float*)(ws + (size_t)N * D_DIM * 4 + (size_t)N * 8);
    float* out     = (float*)d_out;

    hipMemsetAsync(sumexp, 0, (size_t)N * sizeof(float), stream);
    hipMemsetAsync(out, 0, (size_t)out_size * sizeof(float), stream);

    // 1 wave per row -> N waves, 8 waves per 256-thread block.
    const int norm_blocks = (N * 32 + 255) / 256;
    ncl_normalize_kernel<<<norm_blocks, 256, 0, stream>>>(z1, z2, z1h, z2h,
                                                          pos, selfexp, N);

    // One block per 32-row i-block; 8 waves split the j-tiles.
    ncl_sim_kernel<<<N / 32, 256, 0, stream>>>(z1h, z2h, sumexp, N);

    ncl_loss_kernel<<<64, 256, 0, stream>>>(pos, sumexp, selfexp, out, N);
}
